// RWKVTimeMix_18021682774240
// MI455X (gfx1250) — matrix operand order repaired and checked by reading, not compiled, not run on hardware
//
#include <hip/hip_runtime.h>
#include <hip/hip_bf16.h>

// ---------------------------------------------------------------------------
// RWKV TimeMix for MI455X (gfx1250, wave32, WMMA, async LDS copies).
// B=8, T=2048, H=1024  ->  BT = 16384 rows, K = N = 1024.
// ---------------------------------------------------------------------------

#define BATCH   8
#define SEQ     2048
#define HID     1024
#define BT      (BATCH * SEQ)          // 16384

typedef __attribute__((ext_vector_type(16))) __bf16 v16bf;
typedef __attribute__((ext_vector_type(8)))  __bf16 v8bf;
typedef __attribute__((ext_vector_type(8)))  float  v8f;
typedef __attribute__((ext_vector_type(4)))  int    v4i;

// CDNA5 async global->LDS copy path (ASYNCcnt), guarded so the file always
// compiles; falls back to VGPR staging (global_load_b128 + ds_store_b128).
#if __has_builtin(__builtin_amdgcn_global_load_async_to_lds_b128) && \
    __has_builtin(__builtin_amdgcn_s_wait_asynccnt)
#define USE_ASYNC_LDS 1
#else
#define USE_ASYNC_LDS 0
#endif

#if USE_ASYNC_LDS
static __device__ __forceinline__ void cp_async16(void* lds, const void* g) {
    // one 16-byte lane-wise async copy: global -> LDS, tracked by ASYNCcnt
    __builtin_amdgcn_global_load_async_to_lds_b128(
        (__attribute__((address_space(1))) v4i*)g,
        (__attribute__((address_space(3))) v4i*)lds,
        /*offset=*/0, /*cpol=*/0);
}
#endif

// ---- float -> bf16, round-to-nearest-even, pure bit manipulation ----------
static __device__ __forceinline__ __bf16 f2bf(float f) {
    union { float f; unsigned u; } a; a.f = f;
    unsigned r = a.u + 0x7fffu + ((a.u >> 16) & 1u);
    unsigned short h = (unsigned short)(r >> 16);
    __bf16 out;
    __builtin_memcpy(&out, &h, 2);
    return out;
}

// Combine two contiguous 8-half LDS chunks into one 16-half WMMA fragment.
static __device__ __forceinline__ v16bf ld_frag(const __bf16* p0, const __bf16* p1) {
    v8bf a = *(const v8bf*)p0;
    v8bf b = *(const v8bf*)p1;
    return __builtin_shufflevector(a, b, 0,1,2,3,4,5,6,7,8,9,10,11,12,13,14,15);
}

// ---------------------------------------------------------------------------
// Kernel 1: weights fp32 -> bf16 (4 matrices of H*H each).
// ---------------------------------------------------------------------------
__global__ void wcvt_kernel(const float* __restrict__ Wk, const float* __restrict__ Wv,
                            const float* __restrict__ Wr, const float* __restrict__ Wo,
                            __bf16* __restrict__ ok, __bf16* __restrict__ ov,
                            __bf16* __restrict__ orr, __bf16* __restrict__ oo) {
    unsigned g   = blockIdx.x * 256u + threadIdx.x;   // 0 .. 4*1048576-1
    unsigned sel = g >> 20;
    unsigned i   = g & 0xFFFFFu;
    const float* s; __bf16* d;
    if      (sel == 0) { s = Wk; d = ok;  }
    else if (sel == 1) { s = Wv; d = ov;  }
    else if (sel == 2) { s = Wr; d = orr; }
    else               { s = Wo; d = oo;  }
    d[i] = f2bf(s[i]);
}

// ---------------------------------------------------------------------------
// Kernel 2: token shift + mix -> xk/xv/xr in bf16.
// ---------------------------------------------------------------------------
__global__ void mix_kernel(const float* __restrict__ x,
                           const float* __restrict__ tmk, const float* __restrict__ tmv,
                           const float* __restrict__ tmr,
                           __bf16* __restrict__ xk, __bf16* __restrict__ xv,
                           __bf16* __restrict__ xr) {
    size_t g  = (size_t)blockIdx.x * 256u + threadIdx.x;  // 0 .. BT*H-1
    int    h  = (int)(g & (HID - 1));
    size_t bt = g >> 10;
    int    t  = (int)(bt & (SEQ - 1));
    float xc = x[g];
    float xp = (t == 0) ? 0.0f : x[g - HID];
    float mk = tmk[h], mv = tmv[h], mr = tmr[h];
    xk[g] = f2bf(xc * mk + xp * (1.0f - mk));
    xv[g] = f2bf(xc * mv + xp * (1.0f - mv));
    xr[g] = f2bf(xc * mr + xp * (1.0f - mr));
}

// ---------------------------------------------------------------------------
// Kernel 3: bf16 GEMM  C[m,n] = sum_k A[m,k] * W[n,k]   (A:[BT,K], W:[N,K])
//   block tile 128x128, 256 threads = 8 waves (wave32),
//   wave grid 4(M) x 2(N); per wave: 2x4 tiles of 16x16 WMMA,
//   K staged through LDS in chunks of 32 (row stride 40 halves: conflict-free).
//   Async path: double-buffered LDS, GLOBAL_LOAD_ASYNC_TO_LDS_B128 + ASYNCcnt.
//   mode == 1 -> sigmoid epilogue (r projection).
// ---------------------------------------------------------------------------
__global__ __launch_bounds__(256) void gemm_bf16_kernel(const __bf16* __restrict__ A,
                                                        const __bf16* __restrict__ W,
                                                        float* __restrict__ C,
                                                        int mode) {
#if USE_ASYNC_LDS
    __shared__ __align__(16) __bf16 As[2][128][40];
    __shared__ __align__(16) __bf16 Bs[2][128][40];
#else
    __shared__ __align__(16) __bf16 As[1][128][40];
    __shared__ __align__(16) __bf16 Bs[1][128][40];
#endif

    const int tid   = threadIdx.x;
    const int mBase = blockIdx.x * 128;
    const int nBase = blockIdx.y * 128;
    const int K = HID, N = HID;

    const int lane = tid & 31;
    const int wave = tid >> 5;          // 0..7
    const int wm   = wave >> 1;         // 0..3 -> 32 rows each
    const int wn   = wave & 1;          // 0..1 -> 64 cols each
    const int hi   = lane >> 4;         // half-wave select
    const int lm   = lane & 15;         // row / col within 16x16 tile

    // staging: 512 16B chunks per tile, 2 per thread (rows 0..63 and 64..127)
    const int ar0 = tid >> 2;           // 0..63
    const int ar1 = ar0 + 64;           // 64..127
    const int ac  = (tid & 3) * 8;      // half offset within 32-wide K chunk

    const __bf16* Ag = A + (size_t)mBase * K;
    const __bf16* Wg = W + (size_t)nBase * K;
    const int NK = K / 32;              // 32 K-chunks

    v8f acc[2][4];
#pragma unroll
    for (int mt = 0; mt < 2; ++mt)
#pragma unroll
        for (int nt = 0; nt < 4; ++nt)
            acc[mt][nt] = (v8f){0.f,0.f,0.f,0.f,0.f,0.f,0.f,0.f};

    auto compute = [&](int buf) {
        // WMMA fragments per ISA 16-bit layouts (wave32):
        //  A: lane hi*16+m holds K = hi*8+{0..7} and 16+hi*8+{0..7} of row m
        //  B: lane hi*16+n holds K = hi*8+{0..7} and 16+hi*8+{0..7} of column n (= W row n)
        v16bf af[2], bfr[4];
#pragma unroll
        for (int mt = 0; mt < 2; ++mt) {
            const __bf16* rowp = &As[buf][wm * 32 + mt * 16 + lm][0];
            af[mt] = ld_frag(rowp + hi * 8, rowp + 16 + hi * 8);
        }
#pragma unroll
        for (int nt = 0; nt < 4; ++nt) {
            const __bf16* rowp = &Bs[buf][wn * 64 + nt * 16 + lm][0];
            bfr[nt] = ld_frag(rowp + hi * 8, rowp + 16 + hi * 8);
        }
#pragma unroll
        for (int mt = 0; mt < 2; ++mt)
#pragma unroll
            for (int nt = 0; nt < 4; ++nt)
                acc[mt][nt] = __builtin_amdgcn_wmma_f32_16x16x32_bf16(
                    false, af[mt], false, bfr[nt], (short)0, acc[mt][nt],
                    false, false);
    };

#if USE_ASYNC_LDS
    // --- double-buffered pipeline on ASYNCcnt ------------------------------
    auto issue = [&](int kc, int buf) {
        size_t ko = (size_t)kc * 32 + ac;
        cp_async16(&As[buf][ar0][ac], Ag + (size_t)ar0 * K + ko);
        cp_async16(&As[buf][ar1][ac], Ag + (size_t)ar1 * K + ko);
        cp_async16(&Bs[buf][ar0][ac], Wg + (size_t)ar0 * K + ko);
        cp_async16(&Bs[buf][ar1][ac], Wg + (size_t)ar1 * K + ko);
    };
    issue(0, 0);
    for (int kc = 0; kc < NK - 1; ++kc) {
        issue(kc + 1, (kc + 1) & 1);               // prefetch next chunk
        __builtin_amdgcn_s_wait_asynccnt(4);       // chunk kc's 4 copies done
        __syncthreads();                           // visible to all waves
        compute(kc & 1);
        __syncthreads();                           // reads done before reuse
    }
    __builtin_amdgcn_s_wait_asynccnt(0);
    __syncthreads();
    compute((NK - 1) & 1);
#else
    // --- fallback: VGPR staging, single buffer -----------------------------
    uint4 ra0, ra1, rb0, rb1;
    auto loadG = [&](int kc) {
        size_t ko = (size_t)kc * 32 + ac;
        ra0 = *(const uint4*)(Ag + (size_t)ar0 * K + ko);
        ra1 = *(const uint4*)(Ag + (size_t)ar1 * K + ko);
        rb0 = *(const uint4*)(Wg + (size_t)ar0 * K + ko);
        rb1 = *(const uint4*)(Wg + (size_t)ar1 * K + ko);
    };
    loadG(0);
    for (int kc = 0; kc < NK; ++kc) {
        __syncthreads();
        *(uint4*)&As[0][ar0][ac] = ra0;
        *(uint4*)&As[0][ar1][ac] = ra1;
        *(uint4*)&Bs[0][ar0][ac] = rb0;
        *(uint4*)&Bs[0][ar1][ac] = rb1;
        if (kc + 1 < NK) loadG(kc + 1);
        __syncthreads();
        compute(0);
    }
#endif

    // Epilogue: D layout (f32 16x16): VGPR v, lanes0-15 -> M=v, lanes16-31 -> M=v+8
#pragma unroll
    for (int mt = 0; mt < 2; ++mt) {
#pragma unroll
        for (int nt = 0; nt < 4; ++nt) {
            int col = nBase + wn * 64 + nt * 16 + lm;
#pragma unroll
            for (int v = 0; v < 8; ++v) {
                int row = mBase + wm * 32 + mt * 16 + hi * 8 + v;
                float val = acc[mt][nt][v];
                if (mode) val = 1.0f / (1.0f + __expf(-val));
                C[(size_t)row * N + col] = val;
            }
        }
    }
}

// ---------------------------------------------------------------------------
// Kernel 4: WKV recurrence (log-space, numerically stable), fused r*wkv and
// bf16 downcast. 8192 independent (b,h) scans, coalesced across h.
// ---------------------------------------------------------------------------
__global__ void wkv_kernel(const float* __restrict__ kbuf, const float* __restrict__ vbuf,
                           const float* __restrict__ rbuf,
                           const float* __restrict__ w, const float* __restrict__ u,
                           __bf16* __restrict__ abuf) {
    int g = blockIdx.x * 256 + threadIdx.x;   // 0 .. BATCH*HID-1
    int h = g & (HID - 1);
    int b = g >> 10;
    float wd = w[h], uf = u[h];
    float aa = 0.f, bb = 0.f, pp = -1e38f;
    size_t base = (size_t)b * SEQ * HID + h;
    for (int t = 0; t < SEQ; ++t) {
        size_t idx = base + (size_t)t * HID;
        float kk = kbuf[idx], vv = vbuf[idx], rr = rbuf[idx];
        float ww = uf + kk;
        float p  = fmaxf(pp, ww);
        float e1 = __expf(pp - p), e2 = __expf(ww - p);
        float wkv = (e1 * aa + e2 * vv) / (e1 * bb + e2 + 1e-9f);
        abuf[idx] = f2bf(rr * wkv);
        float ww2 = wd + pp;
        float p2  = fmaxf(ww2, kk);
        float e1b = __expf(ww2 - p2), e2b = __expf(kk - p2);
        aa = e1b * aa + e2b * vv;
        bb = e1b * bb + e2b;
        pp = p2;
    }
}

// ---------------------------------------------------------------------------
// Host side.
// ---------------------------------------------------------------------------
extern "C" void kernel_launch(void* const* d_in, const int* in_sizes, int n_in,
                              void* d_out, int out_size, void* d_ws, size_t ws_size,
                              hipStream_t stream) {
    const float* x   = (const float*)d_in[0];
    const float* td  = (const float*)d_in[1];   // time_decay [H]
    const float* tf  = (const float*)d_in[2];   // time_first [H]
    const float* tmk = (const float*)d_in[3];
    const float* tmv = (const float*)d_in[4];
    const float* tmr = (const float*)d_in[5];
    const float* Wk  = (const float*)d_in[6];
    const float* Wv  = (const float*)d_in[7];
    const float* Wr  = (const float*)d_in[8];
    const float* Wo  = (const float*)d_in[9];
    float* out = (float*)d_out;

    char* ws = (char*)d_ws;
    const size_t SZ_XBF = (size_t)BT * HID * 2;      // 32 MB (bf16 [BT,H])
    const size_t SZ_WBF = (size_t)HID * HID * 2;     //  2 MB
    const size_t SZ_F32 = (size_t)BT * HID * 4;      // 64 MB

    __bf16* xk  = (__bf16*)(ws);
    __bf16* xv  = (__bf16*)(ws + SZ_XBF);
    __bf16* xr  = (__bf16*)(ws + 2 * SZ_XBF);
    __bf16* wkb = (__bf16*)(ws + 3 * SZ_XBF);
    __bf16* wvb = (__bf16*)(ws + 3 * SZ_XBF + SZ_WBF);
    __bf16* wrb = (__bf16*)(ws + 3 * SZ_XBF + 2 * SZ_WBF);
    __bf16* wob = (__bf16*)(ws + 3 * SZ_XBF + 3 * SZ_WBF);
    float*  kb  = (float*)(ws + 3 * SZ_XBF + 4 * SZ_WBF);
    float*  vb  = (float*)(ws + 3 * SZ_XBF + 4 * SZ_WBF + SZ_F32);
    float*  rb  = (float*)(ws + 3 * SZ_XBF + 4 * SZ_WBF + 2 * SZ_F32);
    __bf16* ab  = xk;   // xk dead after its GEMM; reuse for r*wkv (bf16)

    // 1) weights -> bf16
    wcvt_kernel<<<(4u * HID * HID) / 256u, 256, 0, stream>>>(Wk, Wv, Wr, Wo,
                                                             wkb, wvb, wrb, wob);
    // 2) token mix -> bf16
    mix_kernel<<<((size_t)BT * HID) / 256u, 256, 0, stream>>>(x, tmk, tmv, tmr,
                                                              xk, xv, xr);
    // 3) projections (k, v, r+sigmoid)
    dim3 gg(BT / 128, HID / 128);
    gemm_bf16_kernel<<<gg, 256, 0, stream>>>(xk, wkb, kb, 0);
    gemm_bf16_kernel<<<gg, 256, 0, stream>>>(xv, wvb, vb, 0);
    gemm_bf16_kernel<<<gg, 256, 0, stream>>>(xr, wrb, rb, 1);
    // 4) WKV scan (fused r*wkv, bf16 out)
    wkv_kernel<<<(BATCH * HID) / 256, 256, 0, stream>>>(kb, vb, rb, td, tf, ab);
    // 5) output projection -> d_out
    gemm_bf16_kernel<<<gg, 256, 0, stream>>>(ab, wob, out, 0);
}
